// TemporalGNN_57612691309354
// MI455X (gfx1250) — matrix operand order repaired and checked by reading, not compiled, not run on hardware
//
#include <hip/hip_runtime.h>

// ---------------------------------------------------------------------------
// MI455X reasoning:
//  - Output is E*NC fp32 = 4 GB -> 23.3 TB/s HBM gives a ~172us floor.
//  - GEMM2 (1M x 64 x 1000) = 128 GFLOP: f32 WMMA (16x16x4) would be ~400us
//    (above the floor); f16 WMMA (16x16x32, fp32 accum) is ~50us (below it).
//    => v_wmma_f32_16x16x32_f16 for both MLP layers, fp32 epilogue.
//  - Hn (12.8MB f16) + W1t/W2t (~150KB) must stay resident in the 192MB L2
//    for the 1M random 128B gathers. The 4GB output stream would evict them
//    under default (RT) store policy -> store output with TH=NT
//    (__builtin_nontemporal_store) so the stream rinses through L2.
//  - One-shot streaming inputs (x, edge_attr, edge_index) use NT loads.
// ---------------------------------------------------------------------------

typedef __attribute__((ext_vector_type(16))) _Float16     v16h;
typedef __attribute__((ext_vector_type(8)))  float        v8f;
typedef __attribute__((ext_vector_type(4)))  unsigned int v4u;

#define F_IN   32
#define HID    64
#define EF     8
#define NC     1000
#define NC_PAD 1008   // 63 tiles of 16
#define K1     160    // emb width 136 padded to 5 * 32

__device__ __forceinline__ float sigmoidf_(float x) {
    return 1.0f / (1.0f + __expf(-x));
}

// ----- prep: W1 [136,64] row-major  ->  W1t [64][160] f16, K-major, 0-padded
__global__ void prep_w1_kernel(const float* __restrict__ W1,
                               _Float16* __restrict__ W1t) {
    int i = blockIdx.x * 256 + threadIdx.x;
    if (i >= 64 * K1) return;
    int n = i / K1, k = i % K1;
    float v = (k < 2 * HID + EF) ? W1[k * 64 + n] : 0.0f;
    W1t[n * K1 + k] = (_Float16)v;
}

// ----- prep: W2 [64,1000] row-major -> W2t [1008][64] f16, K-major, 0-padded
__global__ void prep_w2_kernel(const float* __restrict__ W2,
                               _Float16* __restrict__ W2t) {
    int i = blockIdx.x * 256 + threadIdx.x;
    if (i >= NC_PAD * 64) return;
    int c = i / 64, k = i % 64;
    float v = (c < NC) ? W2[k * NC + c] : 0.0f;
    W2t[c * 64 + k] = (_Float16)v;
}

// ----- GCLSTM step with H=C=0:  Hn = sigm(xg3+bc3+wc2*Cn+b3) * tanh(Cn),
//       Cn = sigm(xg0+bc0+b0) * tanh(xg2+bc2+b2),  xg = x @ Wx
__global__ __launch_bounds__(256) void node_fwd_kernel(
    const float* __restrict__ x, const float* __restrict__ Wx,
    const float* __restrict__ b_cheb, const float* __restrict__ wc,
    const float* __restrict__ b, _Float16* __restrict__ Hn, int N) {
    __shared__ float xs[4][F_IN];
    int l = threadIdx.x >> 6;          // node slot in block (4 nodes/block)
    int h = threadIdx.x & 63;          // hidden unit
    int n = blockIdx.x * 4 + l;
    if (h < F_IN && n < N)             // x is streamed once -> NT load
        xs[l][h] = __builtin_nontemporal_load(&x[(size_t)n * F_IN + h]);
    __syncthreads();
    if (n >= N) return;

    float a0 = 0.f, a2 = 0.f, a3 = 0.f;
#pragma unroll
    for (int k = 0; k < F_IN; k++) {
        float xv = xs[l][k];
        a0 = fmaf(xv, Wx[0 * F_IN * HID + k * HID + h], a0);
        a2 = fmaf(xv, Wx[2 * F_IN * HID + k * HID + h], a2);
        a3 = fmaf(xv, Wx[3 * F_IN * HID + k * HID + h], a3);
    }
    float I  = sigmoidf_(a0 + b_cheb[0 * HID + h] + b[0 * HID + h]);
    float T  = tanhf(a2 + b_cheb[2 * HID + h] + b[2 * HID + h]);
    float Cn = I * T;                                   // F*C == 0
    float O  = sigmoidf_(a3 + b_cheb[3 * HID + h] + wc[2 * HID + h] * Cn +
                         b[3 * HID + h]);
    Hn[(size_t)n * HID + h] = (_Float16)(O * tanhf(Cn));
}

// ----- edge MLP: one wave32 per 16-edge tile, 4 waves per block
__global__ __launch_bounds__(128) void edge_mlp_kernel(
    const int* __restrict__ ei, const float* __restrict__ eattr,
    const _Float16* __restrict__ Hn,
    const _Float16* __restrict__ W1t, const float* __restrict__ b1,
    const _Float16* __restrict__ W2t, const float* __restrict__ b2,
    float* __restrict__ out, int E, int ntiles) {
    __shared__ _Float16 embS[4][16 * K1];    // 16 edges x 160 halfs per wave
    __shared__ _Float16 hmidS[4][16 * HID];  // 16 edges x 64 halfs per wave

    int lane = threadIdx.x & 31;
    int wv   = threadIdx.x >> 5;
    int tile = blockIdx.x * 4 + wv;
    bool active = tile < ntiles;

    _Float16* emb  = embS[wv];
    _Float16* hmid = hmidS[wv];
    int r  = lane & 15;   // row / M / N index within fragment
    int hi = lane >> 4;   // half-wave selector

    // ---- gather: lanes (r,hi=0) copy Hn[src], (r,hi=1) copy Hn[dst]+attr+pad
    if (active) {
        int e  = tile * 16 + r;
        int ec = (e < E) ? e : (E - 1);
        int idx = __builtin_nontemporal_load(hi ? &ei[E + ec] : &ei[ec]);
        const v4u* srcp = (const v4u*)(Hn + (size_t)idx * HID);  // L2-resident
        v4u* dstp = (v4u*)(emb + r * K1 + hi * HID);
#pragma unroll
        for (int j = 0; j < 8; j++) dstp[j] = srcp[j];
        if (hi) {
            const float* ea = eattr + (size_t)ec * EF;   // streamed once -> NT
            _Float16* rp = emb + r * K1;
#pragma unroll
            for (int j = 0; j < EF; j++)
                rp[2 * HID + j] = (_Float16)__builtin_nontemporal_load(&ea[j]);
            v4u z = {0u, 0u, 0u, 0u};
            v4u* zp = (v4u*)(rp + 136);   // byte offset 272, 16B aligned
            zp[0] = z; zp[1] = z; zp[2] = z;
        }
    }
    __syncthreads();

    // ---- layer 1: hmid[16x64] = relu(emb[16x160] @ W1t^T + b1), via WMMA
    if (active) {
        v16h A[5];
#pragma unroll
        for (int kf = 0; kf < 5; kf++) {
            // A k order used here (contiguous, same as B): lane(m=r, hi): K = kf*32 + hi*16 + {0..15}
            const _Float16* p = emb + r * K1 + kf * 32 + hi * 16;
            v4u* ap = (v4u*)&A[kf];
            ap[0] = *(const v4u*)(p);
            ap[1] = *(const v4u*)(p + 8);
        }
#pragma unroll
        for (int t = 0; t < 4; t++) {
            v8f acc = {};
#pragma unroll
            for (int kf = 0; kf < 5; kf++) {
                // B layout: lane(n=r, hi): 16 contiguous K-halfs of column n
                const _Float16* q = W1t + (t * 16 + r) * K1 + kf * 32 + hi * 16;
                v16h Bf = *(const v16h*)q;   // 32B aligned, L2-hot
                acc = __builtin_amdgcn_wmma_f32_16x16x32_f16(
                    false, A[kf], false, Bf, (short)0, acc, false, false);
            }
            float bias = b1[t * 16 + r];
#pragma unroll
            for (int i = 0; i < 8; i++) {
                float v = acc[i] + bias;
                v = v > 0.f ? v : 0.f;
                int m = i + hi * 8;          // C/D layout: M = i + 8*hi, N = r
                hmid[m * HID + t * 16 + r] = (_Float16)v;
            }
        }
    }
    __syncthreads();

    // ---- layer 2: out[16x1000] = hmid[16x64] @ W2t^T + b2, via WMMA
    if (active) {
        v16h A2[2];
#pragma unroll
        for (int kf = 0; kf < 2; kf++) {
            const _Float16* p = hmid + r * HID + kf * 32 + hi * 16;
            v4u* ap = (v4u*)&A2[kf];
            ap[0] = *(const v4u*)(p);
            ap[1] = *(const v4u*)(p + 8);
        }
        int e_base = tile * 16;
        bool full = (e_base + 16 <= E);
        for (int ct = 0; ct < NC_PAD / 16; ct++) {
            int c = ct * 16 + r;
            v8f acc = {};
#pragma unroll
            for (int kf = 0; kf < 2; kf++) {
                const _Float16* q = W2t + (size_t)c * HID + kf * 32 + hi * 16;
                v16h Bf = *(const v16h*)q;   // 32B aligned, L2-hot
                acc = __builtin_amdgcn_wmma_f32_16x16x32_f16(
                    false, A2[kf], false, Bf, (short)0, acc, false, false);
            }
            if (c < NC) {
                float bias = b2[c];
#pragma unroll
                for (int i = 0; i < 8; i++) {
                    int m = i + hi * 8;
                    if (full || e_base + m < E) {
                        // 4GB output stream: TH=NT so it does not evict the
                        // L2-resident Hn gather table / weights.
                        __builtin_nontemporal_store(
                            acc[i] + bias,
                            &out[(size_t)(e_base + m) * NC + c]);
                    }
                }
            }
        }
    }
}

extern "C" void kernel_launch(void* const* d_in, const int* in_sizes, int n_in,
                              void* d_out, int out_size, void* d_ws, size_t ws_size,
                              hipStream_t stream) {
    const float* x      = (const float*)d_in[0];
    const int*   ei     = (const int*)d_in[1];
    const float* eattr  = (const float*)d_in[2];
    const float* Wx     = (const float*)d_in[3];
    /* d_in[4] = Th is unused: H starts at zero so H @ Th == 0 */
    const float* b_cheb = (const float*)d_in[5];
    const float* wc     = (const float*)d_in[6];
    const float* b      = (const float*)d_in[7];
    const float* W1     = (const float*)d_in[8];
    const float* b1     = (const float*)d_in[9];
    const float* W2     = (const float*)d_in[10];
    const float* b2     = (const float*)d_in[11];
    float* out = (float*)d_out;

    int N = in_sizes[0] / F_IN;
    int E = in_sizes[1] / 2;

    // workspace carve (all chunks 32B aligned)
    char* ws = (char*)d_ws;
    _Float16* Hn = (_Float16*)ws;
    size_t off = (size_t)N * HID * sizeof(_Float16);      // N*128 bytes
    _Float16* W1t = (_Float16*)(ws + off);
    off += 64 * K1 * sizeof(_Float16);                    // 20480 bytes
    _Float16* W2t = (_Float16*)(ws + off);                // 129024 bytes

    prep_w1_kernel<<<(64 * K1 + 255) / 256, 256, 0, stream>>>(W1, W1t);
    prep_w2_kernel<<<(NC_PAD * 64 + 255) / 256, 256, 0, stream>>>(W2, W2t);
    node_fwd_kernel<<<(N + 3) / 4, 256, 0, stream>>>(x, Wx, b_cheb, wc, b, Hn, N);

    int ntiles  = (E + 15) / 16;
    int nblocks = (ntiles + 3) / 4;
    edge_mlp_kernel<<<nblocks, 128, 0, stream>>>(ei, eattr, Hn, W1t, b1,
                                                 W2t, b2, out, E, ntiles);
}
